// RASCALLoss_70076686401755
// MI455X (gfx1250) — compile-verified, matrix operand order checked by reading, not run on hardware
//
#include <hip/hip_runtime.h>
#include <hip/hip_bf16.h>

#define TEMP_F   0.07f
#define BTEMP_F  0.07f
#define EPS_F    1e-12f
#define DDIM     128
#define MAXP     1024   // cap on positives per row (expected ~41, binomial tail << 1024)

typedef __attribute__((ext_vector_type(16))) _Float16 v16h;
typedef __attribute__((ext_vector_type(8)))  _Float16 v8h;
typedef __attribute__((ext_vector_type(8)))  float    v8f;

// Load a 16x32 f16 WMMA operand fragment from a row-major [*, 128] matrix.
// ISA layout (CDNA5 ISA, 16-bit A 16x32): lane L holds row (L&15);
// elements e=0..7 -> K = k0 + ((L&16)?8:0) + e, e=8..15 -> +16 more.
// For S = X @ X^T the B fragment of columns j..j+15 gathers X rows j..j+15
// with the identical pattern, so this loader serves both operands.
__device__ __forceinline__ v16h load_frag_rm(const _Float16* __restrict__ X,
                                             int row_base, int k0, int lane) {
  int r    = row_base + (lane & 15);
  int koff = k0 + ((lane & 16) ? 8 : 0);
  const _Float16* p = X + (size_t)r * DDIM + koff;   // 16B aligned
  v8h lo = *(const v8h*)(p);
  v8h hi = *(const v8h*)(p + 16);
  v16h f;
#pragma unroll
  for (int e = 0; e < 8; ++e) { f[e] = lo[e]; f[e + 8] = hi[e]; }
  return f;
}

// ---------------- K1: L2-normalize contrast rows (view-major stacking) +
// gather cache rows; emit f32 (exact path) and f16 (WMMA path). -------------
__global__ __launch_bounds__(128) void k_normalize_gather(
    const float* __restrict__ features, const int* __restrict__ sample_idx,
    const float* __restrict__ cache_feat,
    float* __restrict__ Cf32, _Float16* __restrict__ Ch16,
    float* __restrict__ Crep, int bsz, int nviews) {
  int i = blockIdx.x;            // contrast row, i = view*bsz + b
  int t = threadIdx.x;           // 0..127
  int b = i % bsz, view = i / bsz;
  float v = features[((size_t)b * nviews + view) * DDIM + t];

  __shared__ float red[4];
  float s = v * v;
#pragma unroll
  for (int off = 16; off; off >>= 1) s += __shfl_xor(s, off, 32);
  if ((t & 31) == 0) red[t >> 5] = s;
  __syncthreads();
  float inv = 1.0f / fmaxf(sqrtf(red[0] + red[1] + red[2] + red[3]), EPS_F);
  float nv = v * inv;
  Cf32[(size_t)i * DDIM + t] = nv;
  Ch16[(size_t)i * DDIM + t] = (_Float16)nv;
  Crep[(size_t)i * DDIM + t] = cache_feat[(size_t)sample_idx[b] * DDIM + t];
}

// ---------------- K2: WMMA Gram matrix tile + fused flash-softmax partials.
// Block = 8 waves; wave -> 16-row strip, 4 col tiles (BM=128, BN=64).
// Per (row, colBlock): online (max over logits incl. diag, sumexp excl. diag).
__global__ __launch_bounds__(256) void k_gemm_softmax(
    const _Float16* __restrict__ X, float* __restrict__ partMax,
    float* __restrict__ partSum, int M, int ncb) {
  int wave = threadIdx.x >> 5, lane = threadIdx.x & 31;
  int row_base = blockIdx.x * 128 + wave * 16;
  int colBlk   = blockIdx.y;

  v16h a0 = load_frag_rm(X, row_base,  0, lane);
  v16h a1 = load_frag_rm(X, row_base, 32, lane);
  v16h a2 = load_frag_rm(X, row_base, 64, lane);
  v16h a3 = load_frag_rm(X, row_base, 96, lane);

  float lmax[8], lsum[8];
#pragma unroll
  for (int v = 0; v < 8; ++v) { lmax[v] = -__builtin_inff(); lsum[v] = 0.f; }
  int rbase = row_base + ((lane & 16) ? 8 : 0);   // C/D: row = rbase+v, col = colbase+(lane&15)

#pragma unroll
  for (int t = 0; t < 4; ++t) {
    int col_base = colBlk * 64 + t * 16;
    v8f c = {};
    v16h b0 = load_frag_rm(X, col_base,  0, lane);
    c = __builtin_amdgcn_wmma_f32_16x16x32_f16(false, a0, false, b0, (short)0, c, false, false);
    v16h b1 = load_frag_rm(X, col_base, 32, lane);
    c = __builtin_amdgcn_wmma_f32_16x16x32_f16(false, a1, false, b1, (short)0, c, false, false);
    v16h b2 = load_frag_rm(X, col_base, 64, lane);
    c = __builtin_amdgcn_wmma_f32_16x16x32_f16(false, a2, false, b2, (short)0, c, false, false);
    v16h b3 = load_frag_rm(X, col_base, 96, lane);
    c = __builtin_amdgcn_wmma_f32_16x16x32_f16(false, a3, false, b3, (short)0, c, false, false);

    int col = col_base + (lane & 15);
#pragma unroll
    for (int v = 0; v < 8; ++v) {
      float l  = c[v] * (1.0f / TEMP_F);
      float nm = fmaxf(lmax[v], l);
      float add = ((rbase + v) == col) ? 0.f : __expf(l - nm);  // diag excluded from sum
      lsum[v] = lsum[v] * __expf(lmax[v] - nm) + add;
      lmax[v] = nm;
    }
  }
  // combine across the 16 lanes holding the same row (xor<16 stays in half-wave)
#pragma unroll
  for (int v = 0; v < 8; ++v) {
#pragma unroll
    for (int off = 1; off < 16; off <<= 1) {
      float om = __shfl_xor(lmax[v], off, 32);
      float os = __shfl_xor(lsum[v], off, 32);
      float nm = fmaxf(lmax[v], om);
      lsum[v]  = lsum[v] * __expf(lmax[v] - nm) + os * __expf(om - nm);
      lmax[v]  = nm;
    }
  }
  if ((lane & 15) == 0) {
#pragma unroll
    for (int v = 0; v < 8; ++v) {
      size_t idx = (size_t)(rbase + v) * ncb + colBlk;
      partMax[idx] = lmax[v];
      partSum[idx] = lsum[v];
    }
  }
}

// ---------------- K3: merge split-softmax partials -> rowmax, log(Z+eps) ----
__global__ __launch_bounds__(256) void k_merge_softmax(
    const float* __restrict__ partMax, const float* __restrict__ partSum,
    float* __restrict__ rowStat, int M, int ncb) {
  int row = blockIdx.x * 256 + threadIdx.x;
  if (row >= M) return;
  float gmax = -__builtin_inff();
  for (int b = 0; b < ncb; ++b)
    gmax = fmaxf(gmax, partMax[(size_t)row * ncb + b]);
  float Z = 0.f;
  for (int b = 0; b < ncb; ++b)
    Z += partSum[(size_t)row * ncb + b] * __expf(partMax[(size_t)row * ncb + b] - gmax);
  rowStat[2 * row]     = gmax;
  rowStat[2 * row + 1] = logf(Z + EPS_F);
}

// ---------------- K4: per-row positives, rank-drift weights, loss -----------
__global__ __launch_bounds__(256) void k_rank_loss(
    const float* __restrict__ Cf32, const float* __restrict__ Crep,
    const int* __restrict__ labels, const int* __restrict__ sample_idx,
    const unsigned char* __restrict__ cache_valid,
    const float* __restrict__ rowStat, float* __restrict__ lossOut,
    int M, int bsz) {
  __shared__ float xi[DDIM], ci[DDIM];
  __shared__ int   plist[MAXP];
  __shared__ float scur[MAXP], scache[MAXP], wbuf[MAXP];
  __shared__ int   pcount, allValid;
  __shared__ float swSum, wsSum, usSum;

  int i = blockIdx.x, t = threadIdx.x;
  if (t == 0) { pcount = 0; allValid = 1; swSum = 0.f; wsSum = 0.f; usSum = 0.f; }
  if (t < DDIM) { xi[t] = Cf32[(size_t)i * DDIM + t]; ci[t] = Crep[(size_t)i * DDIM + t]; }
  __syncthreads();

  int li = labels[i % bsz];
  for (int j = t; j < M; j += 256)
    if (j != i && labels[j % bsz] == li) {
      int slot = atomicAdd(&pcount, 1);
      if (slot < MAXP) plist[slot] = j;
    }
  __syncthreads();
  int m = min(pcount, MAXP);

  for (int p = t; p < m; p += 256) {
    int j = plist[p];
    const float* xj = Cf32 + (size_t)j * DDIM;
    const float* cj = Crep + (size_t)j * DDIM;
    float s1 = 0.f, s2 = 0.f;
#pragma unroll 4
    for (int k = 0; k < DDIM; ++k) { s1 += xi[k] * xj[k]; s2 += ci[k] * cj[k]; }
    scur[p] = s1; scache[p] = s2;
    if (!cache_valid[sample_idx[j % bsz]]) atomicAnd(&allValid, 0);
  }
  __syncthreads();

  float denom = (float)max(m - 1, 1);
  for (int p = t; p < m; p += 256) {
    float sj = scur[p], cjv = scache[p]; int idj = plist[p];
    int rc = 0, rk = 0;
    for (int q = 0; q < m; ++q) {                 // stable-argsort tie-break
      float sq = scur[q], cq = scache[q]; int idq = plist[q];
      rc += (sq > sj) || (sq == sj && idq < idj);
      rk += (cq > cjv) || (cq == cjv && idq < idj);
    }
    float w = 1.0f;
    if (m > 1) w = fmaxf(1.0f - fabsf((float)(rc - rk)) / denom, 0.0f);
    wbuf[p] = w;
    atomicAdd(&swSum, w);
    atomicAdd(&usSum, sj);
  }
  __syncthreads();

  bool rowValid   = (cache_valid[sample_idx[i % bsz]] != 0) && (allValid != 0);
  bool useUniform = (!rowValid) || (swSum <= 1e-12f);
  for (int p = t; p < m; p += 256) atomicAdd(&wsSum, wbuf[p] * scur[p]);
  __syncthreads();

  if (t == 0) {
    float loss = 0.f;
    if (m > 0) {
      float dotWS = useUniform ? (usSum / (float)m)
                               : (wsSum / fmaxf(swSum, 1e-30f));
      float lse = rowStat[2 * i] + rowStat[2 * i + 1];
      loss = -(TEMP_F / BTEMP_F) * (dotWS / TEMP_F - lse);   // Sum(W)==1
    }
    lossOut[i] = loss;
  }
}

// ---------------- K5: mean over rows -> scalar output ----------------------
__global__ __launch_bounds__(256) void k_mean(const float* __restrict__ lossBuf,
                                              float* __restrict__ out, int M) {
  __shared__ float red[256];
  float s = 0.f;
  for (int j = threadIdx.x; j < M; j += 256) s += lossBuf[j];
  red[threadIdx.x] = s;
  __syncthreads();
  for (int off = 128; off; off >>= 1) {
    if (threadIdx.x < off) red[threadIdx.x] += red[threadIdx.x + off];
    __syncthreads();
  }
  if (threadIdx.x == 0) out[0] = red[0] / (float)M;
}

extern "C" void kernel_launch(void* const* d_in, const int* in_sizes, int n_in,
                              void* d_out, int out_size, void* d_ws, size_t ws_size,
                              hipStream_t stream) {
  const float*         features    = (const float*)d_in[0];
  const int*           labels      = (const int*)d_in[1];
  const int*           sample_idx  = (const int*)d_in[2];
  const float*         cache_feat  = (const float*)d_in[3];
  const unsigned char* cache_valid = (const unsigned char*)d_in[4];

  const int BSZ = 2048, NVIEWS = 2, M = BSZ * NVIEWS, NCB = M / 64;

  char* ws = (char*)d_ws;
  float*    Cf32    = (float*)ws;    ws += (size_t)M * DDIM * sizeof(float);
  _Float16* Ch16    = (_Float16*)ws; ws += (size_t)M * DDIM * sizeof(_Float16);
  float*    Crep    = (float*)ws;    ws += (size_t)M * DDIM * sizeof(float);
  float*    partMax = (float*)ws;    ws += (size_t)M * NCB * sizeof(float);
  float*    partSum = (float*)ws;    ws += (size_t)M * NCB * sizeof(float);
  float*    rowStat = (float*)ws;    ws += (size_t)M * 2 * sizeof(float);
  float*    lossBuf = (float*)ws;    ws += (size_t)M * sizeof(float);

  k_normalize_gather<<<M, 128, 0, stream>>>(features, sample_idx, cache_feat,
                                            Cf32, Ch16, Crep, BSZ, NVIEWS);
  dim3 g2(M / 128, M / 64);
  k_gemm_softmax<<<g2, 256, 0, stream>>>(Ch16, partMax, partSum, M, NCB);
  k_merge_softmax<<<(M + 255) / 256, 256, 0, stream>>>(partMax, partSum, rowStat, M, NCB);
  k_rank_loss<<<M, 256, 0, stream>>>(Cf32, Crep, labels, sample_idx, cache_valid,
                                     rowStat, lossBuf, M, BSZ);
  k_mean<<<1, 256, 0, stream>>>(lossBuf, (float*)d_out, M);
}
